// OutlierQuantLinear_24721831756577
// MI455X (gfx1250) — matrix operand order repaired and checked by reading, not compiled, not run on hardware
//
#include <hip/hip_runtime.h>

typedef __attribute__((ext_vector_type(16))) _Float16 v16h;
typedef __attribute__((ext_vector_type(2)))  _Float16 v2h;
typedef __attribute__((ext_vector_type(8)))  float    v8f;

#define IN_F     4096
#define OUT_F    4096
#define M_ROWS   1024
#define N_W      (IN_F * OUT_F)
#define N_PACKED (N_W / 2)

#define BM 128
#define BN 128
#define BK 32
#define KT (IN_F / BK)          // 128 k-steps
#define SMEM_HALF 16384         // per buffer: A 8KB | B 8KB
// Fragment-order global layouts: one 16x32 (or 32x16) f16 tile = 32 lanes * 32B = 1KB,
// tiles stored [row_tile][k_tile] so k_tile stride = 1KB.

// ---- 1) int4 group dequant -> f16 W in B-fragment order -------------------
// B 32x16 order used here (not ISA table order): lane = (k16half<<4)|(n&15); f16 slot j = k&15 (dword v=(k&15)>>1, half=k&1)
__global__ __launch_bounds__(256) void k_dequant(const int* __restrict__ packed,
                                                 const float* __restrict__ scales,
                                                 const float* __restrict__ offsets,
                                                 _Float16* __restrict__ Wf) {
  int i = blockIdx.x * blockDim.x + threadIdx.x;
  if (i >= N_PACKED) return;
  int b  = packed[i];
  int p0 = i << 1;                    // even; both nibbles same group, same dword slot
  int g  = p0 >> 7;
  float s = scales[g];
  float o = offsets[g];
  int oo = p0 >> 12;                  // output feature n
  int k  = p0 & 4095;                 // even
  int kl = k & 31;
  int lane = (((kl >> 4) & 1) << 4) | (oo & 15);
  size_t dst = ((size_t)((oo >> 4) * KT + (k >> 5)) * 32 + lane) * 16 + (kl & 15);
  v2h w;
  w.x = (_Float16)((float)(b & 15) * s + o);
  w.y = (_Float16)((float)((b >> 4) & 15) * s + o);
  *(v2h*)(Wf + dst) = w;              // j even -> 4B aligned
}

// ---- 2) outlier scatter into B-fragment order -----------------------------
__global__ __launch_bounds__(256) void k_outliers(const int* __restrict__ idx,
                                                  const float* __restrict__ val,
                                                  _Float16* __restrict__ Wf, int n) {
  int i = blockIdx.x * blockDim.x + threadIdx.x;
  if (i >= n) return;
  int p  = idx[i];
  int oo = p >> 12;
  int k  = p & 4095;
  int kl = k & 31;
  int lane = (((kl >> 4) & 1) << 4) | (oo & 15);
  size_t dst = ((size_t)((oo >> 4) * KT + (k >> 5)) * 32 + lane) * 16 + (kl & 15);
  Wf[dst] = (_Float16)val[i];
}

// ---- 3) x fp32 -> f16 in A-fragment order ---------------------------------
// A 16x32 order used here (not ISA table order): lanes 0-15 hold K 0..15; lanes 16-31 hold K 16..31
// lane = (((k>>4)&1)<<4)|(m&15); f16 slot j = k&15
__global__ __launch_bounds__(256) void k_xcvt(const float* __restrict__ x,
                                              _Float16* __restrict__ Af) {
  int i = blockIdx.x * blockDim.x + threadIdx.x;
  if (i >= (M_ROWS * IN_F) / 2) return;
  int p = i << 1;
  float2 v = ((const float2*)x)[i];
  int m = p >> 12;
  int k = p & 4095;                   // even
  int kl = k & 31;
  int lane = (((kl >> 4) & 1) << 4) | (m & 15);
  int j = kl & 15;
  size_t dst = ((size_t)((m >> 4) * KT + (k >> 5)) * 32 + lane) * 16 + j;
  v2h h;
  h.x = (_Float16)v.x;
  h.y = (_Float16)v.y;
  *(v2h*)(Af + dst) = h;              // j even -> 4B aligned
}

// ---- 4) WMMA GEMM with async global->LDS staging --------------------------
__global__ __launch_bounds__(256) void k_gemm(const _Float16* __restrict__ Af,
                                              const _Float16* __restrict__ Bf,
                                              const float* __restrict__ bias,
                                              float* __restrict__ out) {
  __shared__ __align__(1024) unsigned char smem[2 * SMEM_HALF];

  const int tid  = threadIdx.x;
  const int lane = tid & 31;
  const int wave = tid >> 5;
  const int wm   = wave & 3;          // 4 waves along M: 32 rows each
  const int wn   = wave >> 2;         // 2 waves along N: 64 cols each
  const int bm   = blockIdx.y * BM;
  const int bn   = blockIdx.x * BN;

  const unsigned ldsBase = (unsigned)(size_t)&smem[0];

  // Per-thread staging chunks (16B each): A tile 8KB -> 2 chunks, B tile 8KB -> 2 chunks.
  unsigned long long aG[2], bG[2];
  unsigned aL[2], bL[2];
#pragma unroll
  for (int it = 0; it < 2; ++it) {
    int c    = tid + it * 256;        // 512 chunks
    int t    = c >> 6;                // tile 0..7
    int rest = (c & 63) << 4;         // byte offset in 1KB tile
    aG[it] = (unsigned long long)(size_t)Af + (((size_t)((bm >> 4) + t) * KT) << 10) + rest;
    bG[it] = (unsigned long long)(size_t)Bf + (((size_t)((bn >> 4) + t) * KT) << 10) + rest;
    aL[it] = (unsigned)(t * 1024 + rest);
    bL[it] = (unsigned)(8192 + t * 1024 + rest);
  }

  auto issue = [&](int kt, int buf) {
    const unsigned lb = ldsBase + (unsigned)buf * SMEM_HALF;
    const unsigned long long ko = (unsigned long long)kt << 10;   // k_tile stride = 1KB
#pragma unroll
    for (int it = 0; it < 2; ++it) {
      asm volatile("global_load_async_to_lds_b128 %0, %1, off"
                   :: "v"(lb + aL[it]), "v"(aG[it] + ko) : "memory");
      asm volatile("global_load_async_to_lds_b128 %0, %1, off"
                   :: "v"(lb + bL[it]), "v"(bG[it] + ko) : "memory");
    }
  };

  v8f acc[2][4] = {};

  issue(0, 0);
  int buf = 0;
  for (int kt = 0; kt < KT; ++kt) {
    const bool more = (kt + 1 < KT);
    if (more) {
      issue(kt + 1, buf ^ 1);
      asm volatile("s_wait_asynccnt 0x4" ::: "memory");  // previous group landed
    } else {
      asm volatile("s_wait_asynccnt 0x0" ::: "memory");
    }
    __syncthreads();

    const _Float16* Sb = (const _Float16*)(smem + buf * SMEM_HALF);
    v16h af[2], bf[4];
#pragma unroll
    for (int mt = 0; mt < 2; ++mt)
      af[mt] = *(const v16h*)(Sb + ((wm * 2 + mt) * 512 + lane * 16));
#pragma unroll
    for (int nt = 0; nt < 4; ++nt)
      bf[nt] = *(const v16h*)(Sb + 4096 + ((wn * 4 + nt) * 512 + lane * 16));
#pragma unroll
    for (int mt = 0; mt < 2; ++mt)
#pragma unroll
      for (int nt = 0; nt < 4; ++nt)
        acc[mt][nt] = __builtin_amdgcn_wmma_f32_16x16x32_f16(
            false, af[mt], false, bf[nt], (short)0, acc[mt][nt], false, false);

    __syncthreads();                  // readers done before DMA may overwrite buf
    buf ^= 1;
  }

  // Epilogue: C/D layout -> VGPR r: lanes 0-15 (M=r,N=lane), lanes 16-31 (M=8+r,N=lane-16)
  const int cn = lane & 15;
  const int mo = (lane >> 4) << 3;
#pragma unroll
  for (int nt = 0; nt < 4; ++nt) {
    int col  = bn + (wn * 4 + nt) * 16 + cn;
    float bv = bias[col];
#pragma unroll
    for (int mt = 0; mt < 2; ++mt) {
      int rbase = bm + (wm * 2 + mt) * 16 + mo;
#pragma unroll
      for (int r = 0; r < 8; ++r)
        out[(size_t)(rbase + r) * OUT_F + col] = acc[mt][nt][r] + bv;
    }
  }
}

extern "C" void kernel_launch(void* const* d_in, const int* in_sizes, int n_in,
                              void* d_out, int out_size, void* d_ws, size_t ws_size,
                              hipStream_t stream) {
  const float* x       = (const float*)d_in[0];
  const int*   packed  = (const int*)d_in[1];
  const float* scales  = (const float*)d_in[2];
  const float* offsets = (const float*)d_in[3];
  const int*   oidx    = (const int*)d_in[4];
  const float* oval    = (const float*)d_in[5];
  const float* bias    = (const float*)d_in[6];
  float*       out     = (float*)d_out;
  const int n_outliers = in_sizes[4];

  _Float16* Wf = (_Float16*)d_ws;                              // 33,554,432 B
  _Float16* Xf = (_Float16*)((char*)d_ws + (size_t)N_W * 2);   //  8,388,608 B

  k_dequant<<<N_PACKED / 256, 256, 0, stream>>>(packed, scales, offsets, Wf);
  k_outliers<<<(n_outliers + 255) / 256, 256, 0, stream>>>(oidx, oval, Wf, n_outliers);
  k_xcvt<<<(M_ROWS * IN_F / 2) / 256, 256, 0, stream>>>(x, Xf);

  dim3 grid(OUT_F / BN, M_ROWS / BM);   // (32, 8) = 256 blocks ~ one per WGP
  k_gemm<<<grid, 256, 0, stream>>>(Xf, Wf, bias, out);
}
